// Attention_21079699489228
// MI455X (gfx1250) — matrix operand order repaired and checked by reading, not compiled, not run on hardware
//
#include <hip/hip_runtime.h>
#include <hip/hip_bf16.h>
#include <cstddef>

// ---------------------------------------------------------------------------
// Problem constants (from the reference)
// ---------------------------------------------------------------------------
#define BATCH   32
#define NPIX    4096
#define EDIM    512
#define ADIM    128

typedef __attribute__((ext_vector_type(16))) __bf16          v16bf;
typedef __attribute__((ext_vector_type(16))) unsigned short  v16u;
typedef __attribute__((ext_vector_type(8)))  float           v8f;

// f32 -> bf16 via native conversion (lowers to v_cvt_pk_bf16_f32 on gfx1250)
__device__ __forceinline__ __bf16 f2bf_native(float f) {
    return (__bf16)f;
}

// f32 -> bf16 bits, explicit round-to-nearest-even (used in one-time pack)
__device__ __forceinline__ unsigned short f2bf_bits(float f) {
    unsigned int u = __float_as_uint(f);
    unsigned int r = u + 0x7FFFu + ((u >> 16) & 1u);
    return (unsigned short)(r >> 16);
}

// ---------------------------------------------------------------------------
// Kernel A1: pre-swizzle W_enc [512x128] f32 into WMMA B-fragment order, bf16.
// Bfrag[kt][nt][lane][j] = W_enc[kt*32 + (lane>>4)*16 + j][nt*16 + (lane&15)]
// kt in [0,16), nt in [0,8), lane in [0,32), j in [0,16)  -> 65536 elements.
// ---------------------------------------------------------------------------
__global__ void pack_benc_kernel(const float* __restrict__ W_enc,
                                 unsigned short* __restrict__ bfrag) {
    for (int t = blockIdx.x * blockDim.x + threadIdx.x; t < 16 * 8 * 32 * 16;
         t += gridDim.x * blockDim.x) {
        int j    = t & 15;
        int lane = (t >> 4) & 31;
        int nt   = (t >> 9) & 7;
        int kt   = t >> 12;
        int k = kt * 32 + ((lane >> 4) << 4) + j;
        int n = nt * 16 + (lane & 15);
        bfrag[t] = f2bf_bits(W_enc[k * ADIM + n]);
    }
}

// ---------------------------------------------------------------------------
// Kernel A2: att2[b][a] = decoder_hidden[b,:] . W_dec[:,a] + b_dec[a]
// grid = 32 blocks (one per batch), 128 threads (one per a).
// ---------------------------------------------------------------------------
__global__ void att2_kernel(const float* __restrict__ dec,
                            const float* __restrict__ W_dec,
                            const float* __restrict__ b_dec,
                            float* __restrict__ att2) {
    int b = blockIdx.x;
    int a = threadIdx.x;
    const float* dh = dec + b * EDIM;
    float acc = 0.f;
#pragma unroll 4
    for (int e = 0; e < EDIM; ++e)
        acc = fmaf(dh[e], W_dec[e * ADIM + a], acc);
    att2[b * ADIM + a] = acc + b_dec[a];
}

// ---------------------------------------------------------------------------
// Kernel B: the WMMA GEMM + fused score epilogue.
// grid = (B*N)/128 = 1024 WGs of 256 threads (8 waves).
// Each wave: one 16-row M tile, 8 N-tiles of 16, K loop 16 x (K=32 bf16 WMMA).
// Epilogue: score[row] = sum_a relu((C+b_enc)*att2)*w_full + b_full
// Raw scores are written into the alpha region of d_out (overwritten later).
// ---------------------------------------------------------------------------
__global__ void score_kernel(const float* __restrict__ enc,
                             const unsigned short* __restrict__ bfrag,
                             const float* __restrict__ att2,
                             const float* __restrict__ b_enc,
                             const float* __restrict__ w_full,
                             const float* __restrict__ b_full,
                             float* __restrict__ score_out) {
    const int rows0 = blockIdx.x * 128;          // global row (b*N + n)
    const int b     = rows0 >> 12;               // /4096
    const int n0    = rows0 & (NPIX - 1);
    const int wave  = threadIdx.x >> 5;
    const int lane  = threadIdx.x & 31;
    const int half  = lane >> 4;                 // 0 / 1
    const int l16   = lane & 15;

    const int nrow = n0 + wave * 16 + l16;       // row this lane feeds to A-frag
    const float* rowp = enc + ((size_t)b * NPIX + nrow) * EDIM;

    // per-lane epilogue params for n = nt*16 + l16
    float be[8], a2[8], wf[8];
#pragma unroll
    for (int nt = 0; nt < 8; ++nt) {
        int n = nt * 16 + l16;
        be[nt] = b_enc[n];
        a2[nt] = att2[b * ADIM + n];
        wf[nt] = w_full[n];
    }

    v8f acc[8];
#pragma unroll
    for (int nt = 0; nt < 8; ++nt)
#pragma unroll
        for (int i = 0; i < 8; ++i) acc[nt][i] = 0.f;

    const v16u* bfp = (const v16u*)(bfrag) + lane;   // [kt][nt][lane][16]

    for (int kt = 0; kt < 16; ++kt) {
        // ---- A-fragment: lane half h holds K = {h*16+0..15} ----
        const float* kp = rowp + kt * 32 + half * 16;
        float4 x0 = *(const float4*)(kp);
        float4 x1 = *(const float4*)(kp + 4);
        float4 x2 = *(const float4*)(kp + 8);
        float4 x3 = *(const float4*)(kp + 12);
        v16bf afrag;
        afrag[0]  = f2bf_native(x0.x); afrag[1]  = f2bf_native(x0.y);
        afrag[2]  = f2bf_native(x0.z); afrag[3]  = f2bf_native(x0.w);
        afrag[4]  = f2bf_native(x1.x); afrag[5]  = f2bf_native(x1.y);
        afrag[6]  = f2bf_native(x1.z); afrag[7]  = f2bf_native(x1.w);
        afrag[8]  = f2bf_native(x2.x); afrag[9]  = f2bf_native(x2.y);
        afrag[10] = f2bf_native(x2.z); afrag[11] = f2bf_native(x2.w);
        afrag[12] = f2bf_native(x3.x); afrag[13] = f2bf_native(x3.y);
        afrag[14] = f2bf_native(x3.z); afrag[15] = f2bf_native(x3.w);

        // ---- 8 N-tiles: B-frags are one coalesced 32B load each (L2-hot) ----
#pragma unroll
        for (int nt = 0; nt < 8; ++nt) {
            v16u bu = bfp[(kt * 8 + nt) * 32];
            v16bf bfragv = __builtin_bit_cast(v16bf, bu);
            acc[nt] = __builtin_amdgcn_wmma_f32_16x16x32_bf16(
                false, afrag, false, bfragv, (short)0, acc[nt], false, false);
        }
    }

    // ---- fused epilogue: relu((C+b_enc)*att2)*w_full, reduce over n ----
    const float bf = b_full[0];
#pragma unroll
    for (int i = 0; i < 8; ++i) {
        float s = 0.f;
#pragma unroll
        for (int nt = 0; nt < 8; ++nt) {
            float v = (acc[nt][i] + be[nt]) * a2[nt];
            v = v > 0.f ? v : 0.f;
            s = fmaf(v, wf[nt], s);
        }
        // reduce across the 16 lanes of each half (rows i and 8+i)
        s += __shfl_xor(s, 1);
        s += __shfl_xor(s, 2);
        s += __shfl_xor(s, 4);
        s += __shfl_xor(s, 8);
        if (l16 == 0) {
            int row = n0 + wave * 16 + half * 8 + i;   // C layout: M=(half*8+i)
            score_out[b * NPIX + row] = s + bf;
        }
    }
}

// ---------------------------------------------------------------------------
// Kernel C: masked softmax over N per batch, in place in the alpha region.
// grid = 32 blocks of 256 threads; full row staged in LDS.
// ---------------------------------------------------------------------------
__global__ void softmax_kernel(float* __restrict__ alpha,        // in: scores
                               const float* __restrict__ mask,
                               const int* __restrict__ mask_need) {
    __shared__ float s[NPIX];
    __shared__ float red[256];
    const int b = blockIdx.x;
    const int t = threadIdx.x;
    const int mn = mask_need[0];

    float mx = -3.402823466e38f;
#pragma unroll
    for (int j = 0; j < NPIX / 256; ++j) {
        int n = j * 256 + t;
        float v = alpha[b * NPIX + n];
        if (mn == 1) v *= mask[b * NPIX + n];
        s[n] = v;
        mx = fmaxf(mx, v);
    }
    red[t] = mx;
    __syncthreads();
    for (int off = 128; off > 0; off >>= 1) {
        if (t < off) red[t] = fmaxf(red[t], red[t + off]);
        __syncthreads();
    }
    mx = red[0];
    __syncthreads();

    float sum = 0.f;
#pragma unroll
    for (int j = 0; j < NPIX / 256; ++j) {
        int n = j * 256 + t;
        float e = __expf(s[n] - mx);
        s[n] = e;
        sum += e;
    }
    red[t] = sum;
    __syncthreads();
    for (int off = 128; off > 0; off >>= 1) {
        if (t < off) red[t] += red[t + off];
        __syncthreads();
    }
    const float inv = 1.f / red[0];

#pragma unroll
    for (int j = 0; j < NPIX / 256; ++j) {
        int n = j * 256 + t;
        float a = s[n] * inv;
        if (mn == 1) a *= mask[b * NPIX + n];
        alpha[b * NPIX + n] = a;
    }
}

// ---------------------------------------------------------------------------
// Kernel D: awe[b][e] = sum_n alpha[b][n] * enc[b][n][e]
// grid = 32*4 blocks of 128 threads (one e per thread); alpha in LDS.
// Memory-bound: second full 256 MB stream of encoder_out, coalesced.
// ---------------------------------------------------------------------------
__global__ void awe_kernel(const float* __restrict__ enc,
                           const float* __restrict__ alpha,
                           float* __restrict__ awe) {
    __shared__ float la[NPIX];
    const int b  = blockIdx.x >> 2;
    const int et = blockIdx.x & 3;
    const int t  = threadIdx.x;
    const int e  = et * 128 + t;

#pragma unroll
    for (int j = 0; j < NPIX / 128; ++j)
        la[j * 128 + t] = alpha[b * NPIX + j * 128 + t];
    __syncthreads();

    const float* ep = enc + (size_t)b * NPIX * EDIM + e;
    float acc = 0.f;
#pragma unroll 4
    for (int n = 0; n < NPIX; ++n)
        acc = fmaf(la[n], ep[(size_t)n * EDIM], acc);
    awe[b * EDIM + e] = acc;
}

// ---------------------------------------------------------------------------
// Launcher
// ---------------------------------------------------------------------------
extern "C" void kernel_launch(void* const* d_in, const int* in_sizes, int n_in,
                              void* d_out, int out_size, void* d_ws, size_t ws_size,
                              hipStream_t stream) {
    const float* enc    = (const float*)d_in[0];   // [B,N,512]
    const float* mask   = (const float*)d_in[1];   // [B,N]
    const float* dec    = (const float*)d_in[2];   // [B,512]
    const float* W_enc  = (const float*)d_in[3];   // [512,128]
    const float* b_enc  = (const float*)d_in[4];   // [128]
    const float* W_dec  = (const float*)d_in[5];   // [512,128]
    const float* b_dec  = (const float*)d_in[6];   // [128]
    const float* w_full = (const float*)d_in[7];   // [128]
    const float* b_full = (const float*)d_in[8];   // scalar
    const int*   mneed  = (const int*)d_in[9];     // scalar

    float* out   = (float*)d_out;
    float* awe   = out;                            // [B,512]
    float* alpha = out + BATCH * EDIM;             // [B,N]

    unsigned short* bfrag = (unsigned short*)d_ws;                // 128 KB
    float* att2 = (float*)((char*)d_ws + 16 * 8 * 32 * 16 * 2);   // 16 KB

    pack_benc_kernel<<<64, 256, 0, stream>>>(W_enc, bfrag);
    att2_kernel<<<BATCH, ADIM, 0, stream>>>(dec, W_dec, b_dec, att2);
    score_kernel<<<(BATCH * NPIX) / 128, 256, 0, stream>>>(
        enc, bfrag, att2, b_enc, w_full, b_full, alpha);
    softmax_kernel<<<BATCH, 256, 0, stream>>>(alpha, mask, mneed);
    awe_kernel<<<BATCH * 4, 128, 0, stream>>>(enc, alpha, awe);
}
